// multihead_self_attention_1546188226971
// MI455X (gfx1250) — compile-verified, matrix operand order checked by reading, not run on hardware
//
#include <hip/hip_runtime.h>
#include <hip/hip_bf16.h>

// ---------------------------------------------------------------------------
// Multi-head self-attention forward (B=4, S=2048, D=1024, H=16, d_k=64)
// bf16 WMMA (V_WMMA_F32_16X16X32_BF16), fp32 accumulation, fp32 online
// softmax. Flash-attention streaming over keys with TDM (tensor_load_to_lds)
// double-buffered staging of K/V tiles into LDS.
// ---------------------------------------------------------------------------

typedef __bf16 bf16;
typedef __attribute__((ext_vector_type(16))) __bf16        v16bf;
typedef __attribute__((ext_vector_type(8)))  __bf16        v8bf;
typedef __attribute__((ext_vector_type(4)))  __bf16        v4bf;
typedef __attribute__((ext_vector_type(8)))  float         v8f;
typedef __attribute__((ext_vector_type(4)))  float         v4f;
typedef __attribute__((ext_vector_type(4)))  unsigned int  v4u;
typedef __attribute__((ext_vector_type(8)))  int           v8i;
typedef __attribute__((ext_vector_type(4)))  int           v4i;

#define D_MODEL 1024
#define S_LEN   2048
#define NHEADS  16
#define DK      64
#define BATCH   4
#define MROWS   (BATCH * S_LEN)   /* 8192 */

#if __has_builtin(__builtin_amdgcn_tensor_load_to_lds) && \
    __has_builtin(__builtin_amdgcn_s_wait_tensorcnt)
#define HAVE_TDM 1
#else
#define HAVE_TDM 0
#endif

// ---- float -> bf16 (round to nearest even) ---------------------------------
__device__ __forceinline__ bf16 f2bf(float f) {
  unsigned u = __float_as_uint(f);
  u += 0x7FFFu + ((u >> 16) & 1u);
  unsigned short s = (unsigned short)(u >> 16);
  bf16 r;
  __builtin_memcpy(&r, &s, sizeof(r));
  return r;
}

// ---- bf16 WMMA fragment loader ---------------------------------------------
// 16-bit A (16x32, MxK) / B (32x16, KxN) fragment per ISA layout:
//   lane L: fixed index (M for A, N for B) = L&15 ; h = L>>4
//   elements [0..7]  hold k = h*8 + 0..7
//   elements [8..15] hold k = 16 + h*8 + 0..7
// Row-major storage contiguous along k => two 16-byte loads per lane.
__device__ __forceinline__ v16bf load_frag(const bf16* base, int stride,
                                           int row, int k0, int lane) {
  const int h = lane >> 4;
  const bf16* p = base + (size_t)row * (size_t)stride + k0 + h * 8;
  v8bf lo = *(const v8bf*)(p);
  v8bf hi = *(const v8bf*)(p + 16);
  v16bf f;
#pragma unroll
  for (int i = 0; i < 8; ++i) { f[i] = lo[i]; f[i + 8] = hi[i]; }
  return f;
}

__device__ __forceinline__ v8f wmma_bf(v16bf a, v16bf b, v8f c) {
  return __builtin_amdgcn_wmma_f32_16x16x32_bf16(
      /*neg_a=*/false, a, /*neg_b=*/false, b,
      /*c_mod=*/(short)0, c, /*reuse_a=*/false, /*reuse_b=*/false);
}

// ---- TDM 2D tile load: global -> LDS ---------------------------------------
// Builds the Tensor DMA Descriptor (D#) per CDNA5 ISA 8.3/8.4 and issues
// TENSOR_LOAD_TO_LDS (tracked with TENSORcnt). 2D tile: tile0 elements
// contiguous, tile1 rows with stride0 (data_size = 2 bytes).
// Toolchain here uses the 6-arg builtin: (g0, g1, g2, g3, gx, cpol).
__device__ __forceinline__ void tdm_load_2d(const bf16* gptr, unsigned lds_addr,
                                            unsigned tile0, unsigned tile1,
                                            unsigned stride0) {
#if HAVE_TDM
  const unsigned long long ga = (unsigned long long)(uintptr_t)gptr;
  v4u g0;
  g0[0] = 0x1u;                                           // count=1, user D#
  g0[1] = lds_addr;                                       // LDS byte address
  g0[2] = (unsigned)ga;                                   // global_addr[31:0]
  g0[3] = (unsigned)((ga >> 32) & 0x01FFFFFFu) | (2u << 30); // [56:32] | type=2
  const unsigned tdim0 = tile0;                           // OOB bound = tile
  const unsigned tdim1 = tile1;
  v8i g1;
  g1[0] = (int)(1u << 16);                                // data_size=2B, mask=0
  g1[1] = (int)((tdim0 & 0xFFFFu) << 16);                 // tensor_dim0[15:0]
  g1[2] = (int)(((tdim0 >> 16) & 0xFFFFu) | ((tdim1 & 0xFFFFu) << 16));
  g1[3] = (int)(((tdim1 >> 16) & 0xFFFFu) | ((tile0 & 0xFFFFu) << 16));
  g1[4] = (int)(tile1 & 0xFFFFu);                         // tile_dim1, tile_dim2=0
  g1[5] = (int)stride0;                                   // tensor_dim0_stride lo
  g1[6] = 0;                                              // stride hi / dim1_stride
  g1[7] = 0;
  v4i z4 = {0, 0, 0, 0};
  v8i z8 = {0, 0, 0, 0, 0, 0, 0, 0};
  __builtin_amdgcn_tensor_load_to_lds(g0, g1, z4, z4, z8, 0);
#else
  (void)gptr; (void)lds_addr; (void)tile0; (void)tile1; (void)stride0;
#endif
}

// ---------------------------------------------------------------------------
// Kernel 1: fp32 -> bf16 conversion
// ---------------------------------------------------------------------------
__global__ void __launch_bounds__(256)
cvt_f32_bf16(const float* __restrict__ src, bf16* __restrict__ dst, int n4) {
  int i = blockIdx.x * blockDim.x + threadIdx.x;
  int stride = gridDim.x * blockDim.x;
  for (; i < n4; i += stride) {
    v4f v = ((const v4f*)src)[i];
    v4bf o;
#pragma unroll
    for (int j = 0; j < 4; ++j) o[j] = f2bf(v[j]);
    ((v4bf*)dst)[i] = o;
  }
}

// ---------------------------------------------------------------------------
// Kernel 2: Y = X @ W^T + bias   (X: [8192,1024] bf16, W: [1024,1024] bf16)
// Block = 256 threads = 8 waves; block tile 128x128; wave tile 32x64
// (2 A-frags x 4 B-frags -> 8 WMMAs per K-chunk, 0.75 loads/wmma).
// mode 0: Y bf16 row-major [M][N]
// mode 1: Y bf16 as V^T per head: [(b*H+h)*64+d][s]
// mode 2: Y fp32 row-major [M][N]
// ---------------------------------------------------------------------------
__global__ void __launch_bounds__(256)
gemm_bf16(const bf16* __restrict__ X, const bf16* __restrict__ W,
          const float* __restrict__ bias, void* __restrict__ Y, int mode) {
  const int lane = threadIdx.x & 31;
  const int w    = threadIdx.x >> 5;
  const int nl   = lane & 15;
  const int hh   = lane >> 4;
  const int m0 = blockIdx.x * 128 + (w & 3) * 32;
  const int n0 = blockIdx.y * 128 + (w >> 2) * 64;

  v8f acc[2][4] = {};
#pragma unroll 2
  for (int kc = 0; kc < D_MODEL; kc += 32) {
    v16bf a0 = load_frag(X, D_MODEL, m0 + nl,      kc, lane);
    v16bf a1 = load_frag(X, D_MODEL, m0 + 16 + nl, kc, lane);
    if (kc + 32 < D_MODEL) {
      __builtin_prefetch(X + (size_t)(m0 + nl) * D_MODEL + kc + 32, 0, 3);
      __builtin_prefetch(X + (size_t)(m0 + 16 + nl) * D_MODEL + kc + 32, 0, 3);
    }
#pragma unroll
    for (int t = 0; t < 4; ++t) {
      v16bf b = load_frag(W, D_MODEL, n0 + t * 16 + nl, kc, lane);
      acc[0][t] = wmma_bf(a0, b, acc[0][t]);
      acc[1][t] = wmma_bf(a1, b, acc[1][t]);
    }
  }

#pragma unroll
  for (int i = 0; i < 2; ++i) {
#pragma unroll
    for (int t = 0; t < 4; ++t) {
#pragma unroll
      for (int r = 0; r < 8; ++r) {
        const int m = m0 + i * 16 + r + 8 * hh;
        const int n = n0 + t * 16 + nl;
        const float val = acc[i][t][r] + bias[n];
        if (mode == 0) {
          ((bf16*)Y)[(size_t)m * D_MODEL + n] = f2bf(val);
        } else if (mode == 1) {
          const int b  = m >> 11;
          const int s  = m & (S_LEN - 1);
          const int hd = n >> 6;
          const int d  = n & (DK - 1);
          ((bf16*)Y)[((size_t)((b * NHEADS + hd) * DK + d)) * S_LEN + s] = f2bf(val);
        } else {
          ((float*)Y)[(size_t)m * D_MODEL + n] = val;
        }
      }
    }
  }
}

// ---------------------------------------------------------------------------
// Kernel 3: causal flash attention.
// grid = (S/128, B*H), block = 256 (8 waves). Block owns 128 queries
// (wave w: rows qbase + w*16 ..), all waves share one key stream.
// K (32x64) and V^T (64x32) tiles staged into LDS by the Tensor Data Mover,
// double-buffered: wave 0 issues TDM for block i+1, waits TENSORcnt<=2 for
// block i, workgroup barriers publish the tile to all waves.
// ---------------------------------------------------------------------------
__global__ void __launch_bounds__(256)
attn_fwd(const bf16* __restrict__ qb, const bf16* __restrict__ kb,
         const bf16* __restrict__ vt, bf16* __restrict__ ob) {
  const int lane  = threadIdx.x & 31;
  const int w     = threadIdx.x >> 5;
  const int nl    = lane & 15;
  const int hh    = lane >> 4;
  const int bh    = blockIdx.y;
  const int b     = bh >> 4;
  const int hd    = bh & 15;
  const int qbase = blockIdx.x * 128;
  const int q0    = qbase + w * 16;

  __shared__ __align__(16) bf16 Ktile[2][32][DK];   // 8 KB  [key][d]
  __shared__ __align__(16) bf16 Vtile[2][DK][32];   // 8 KB  [d][key]
  __shared__ __align__(16) bf16 Pbuf[8][16][32];    // 8 KB  per-wave P tile

  const bf16* qbase_p = qb + (size_t)b * S_LEN * D_MODEL;
  const bf16* kbase_p = kb + (size_t)b * S_LEN * D_MODEL;
  const bf16* vbase_p = vt + (size_t)bh * DK * S_LEN;   // V^T rows: [d][s]

  // Q fragments (16x64 = two K=32 chunks), loaded once per wave
  const v16bf qf0 = load_frag(qbase_p, D_MODEL, q0 + nl, hd * DK + 0,  lane);
  const v16bf qf1 = load_frag(qbase_p, D_MODEL, q0 + nl, hd * DK + 32, lane);

  v8f o[4] = {};
  float mrow[8], lrow[8];
#pragma unroll
  for (int r = 0; r < 8; ++r) { mrow[r] = -1e30f; lrow[r] = 0.0f; }

  const float scale = 0.125f;          // 1/sqrt(64)
  const int   nb    = (qbase + 128) >> 5;   // key blocks of 32, causal bound

#if HAVE_TDM
  if (w == 0) {
    tdm_load_2d(kbase_p + (size_t)0 * D_MODEL + hd * DK,
                (unsigned)(uintptr_t)&Ktile[0][0][0], DK, 32, D_MODEL);
    tdm_load_2d(vbase_p + 0,
                (unsigned)(uintptr_t)&Vtile[0][0][0], 32, DK, S_LEN);
  }
#endif

  for (int blk = 0; blk < nb; ++blk) {
    const int buf = blk & 1;
    const int k0  = blk * 32;

#if HAVE_TDM
    if (w == 0) {
      if (blk + 1 < nb) {
        const int kn = (blk + 1) * 32;
        tdm_load_2d(kbase_p + (size_t)kn * D_MODEL + hd * DK,
                    (unsigned)(uintptr_t)&Ktile[buf ^ 1][0][0], DK, 32, D_MODEL);
        tdm_load_2d(vbase_p + kn,
                    (unsigned)(uintptr_t)&Vtile[buf ^ 1][0][0], 32, DK, S_LEN);
        __builtin_amdgcn_s_wait_tensorcnt(2);   // oldest K/V pair complete
      } else {
        __builtin_amdgcn_s_wait_tensorcnt(0);
      }
    }
    __syncthreads();                             // publish tile to all waves
#else
    // Fallback: cooperative synchronous staging (one b128 per thread/tile)
    {
      const int tid = threadIdx.x;
      const int kr = tid >> 3, kc8 = (tid & 7) * 8;       // K: 32 rows x 64
      *(v8bf*)&Ktile[buf][kr][kc8] =
          *(const v8bf*)(kbase_p + (size_t)(k0 + kr) * D_MODEL + hd * DK + kc8);
      const int vr = tid >> 2, vc8 = (tid & 3) * 8;       // V: 64 rows x 32
      *(v8bf*)&Vtile[buf][vr][vc8] =
          *(const v8bf*)(vbase_p + (size_t)vr * S_LEN + k0 + vc8);
    }
    __syncthreads();
#endif

    const bf16* Kt = &Ktile[buf][0][0];
    const bf16* Vt = &Vtile[buf][0][0];

    // ---- S = Q K^T (two 16-key subtiles) ----
    v16bf kf0a = load_frag(Kt, DK, nl,      0,  lane);
    v16bf kf0b = load_frag(Kt, DK, nl,      32, lane);
    v16bf kf1a = load_frag(Kt, DK, nl + 16, 0,  lane);
    v16bf kf1b = load_frag(Kt, DK, nl + 16, 32, lane);

    v8f s0 = {}; v8f s1 = {};
    s0 = wmma_bf(qf0, kf0a, s0);
    s0 = wmma_bf(qf1, kf0b, s0);
    s1 = wmma_bf(qf0, kf1a, s1);
    s1 = wmma_bf(qf1, kf1b, s1);

    // ---- scale, causal mask, online softmax (fp32 stats) ----
    float p0[8], p1[8];
#pragma unroll
    for (int r = 0; r < 8; ++r) {
      const int qrow = q0 + r + 8 * hh;
      const int key0 = k0 + nl;
      const int key1 = key0 + 16;
      float v0 = s0[r] * scale; if (key0 > qrow) v0 = -1e30f;
      float v1 = s1[r] * scale; if (key1 > qrow) v1 = -1e30f;

      float bm = fmaxf(v0, v1);
#pragma unroll
      for (int mk = 1; mk < 16; mk <<= 1)
        bm = fmaxf(bm, __shfl_xor(bm, mk, 32));   // reduce within 16-lane half

      const float mnew  = fmaxf(mrow[r], bm);
      const float alpha = __expf(mrow[r] - mnew);
      const float e0 = __expf(v0 - mnew);
      const float e1 = __expf(v1 - mnew);
      float rs = e0 + e1;
#pragma unroll
      for (int mk = 1; mk < 16; mk <<= 1)
        rs += __shfl_xor(rs, mk, 32);

      lrow[r] = lrow[r] * alpha + rs;
      mrow[r] = mnew;
      p0[r] = e0; p1[r] = e1;
#pragma unroll
      for (int t = 0; t < 4; ++t) o[t][r] *= alpha;
    }

    // ---- restage P (16x32) via LDS into A-fragment order ----
#pragma unroll
    for (int r = 0; r < 8; ++r) {
      const int m = r + 8 * hh;
      Pbuf[w][m][nl]      = f2bf(p0[r]);
      Pbuf[w][m][nl + 16] = f2bf(p1[r]);
    }
    const v16bf pf = load_frag(&Pbuf[w][0][0], 32, nl, 0, lane);

    // ---- O += P @ V ----
#pragma unroll
    for (int t = 0; t < 4; ++t) {
      v16bf vf = load_frag(Vt, 32, t * 16 + nl, 0, lane);
      o[t] = wmma_bf(pf, vf, o[t]);
    }

    __syncthreads();   // all waves done with buf before TDM reuses it
  }

  // ---- epilogue: normalize, write bf16 [b][s][h*64+d] for final GEMM ----
#pragma unroll
  for (int r = 0; r < 8; ++r) {
    const float inv = 1.0f / lrow[r];
    const int m = q0 + r + 8 * hh;
#pragma unroll
    for (int t = 0; t < 4; ++t) {
      const int d = t * 16 + nl;
      ob[((size_t)(b * S_LEN + m)) * D_MODEL + hd * DK + d] = f2bf(o[t][r] * inv);
    }
  }
}

// ---------------------------------------------------------------------------
// Host-side orchestration
// ---------------------------------------------------------------------------
extern "C" void kernel_launch(void* const* d_in, const int* in_sizes, int n_in,
                              void* d_out, int out_size, void* d_ws, size_t ws_size,
                              hipStream_t stream) {
  (void)in_sizes; (void)n_in; (void)out_size; (void)ws_size;

  const float* x  = (const float*)d_in[0];
  const float* Wq = (const float*)d_in[1];
  const float* bq = (const float*)d_in[2];
  const float* Wk = (const float*)d_in[3];
  const float* bk = (const float*)d_in[4];
  const float* Wv = (const float*)d_in[5];
  const float* bv = (const float*)d_in[6];
  const float* Wo = (const float*)d_in[7];
  const float* bo = (const float*)d_in[8];

  char* ws = (char*)d_ws;
  size_t off = 0;
  const size_t XB = (size_t)MROWS * D_MODEL * sizeof(bf16);      // 16 MB
  const size_t WB = (size_t)D_MODEL * D_MODEL * sizeof(bf16);    //  2 MB

  bf16* xbf  = (bf16*)(ws + off); off += XB;
  bf16* wqbf = (bf16*)(ws + off); off += WB;
  bf16* wkbf = (bf16*)(ws + off); off += WB;
  bf16* wvbf = (bf16*)(ws + off); off += WB;
  bf16* wobf = (bf16*)(ws + off); off += WB;
  bf16* qbf  = (bf16*)(ws + off); off += XB;
  bf16* kbf  = (bf16*)(ws + off); off += XB;
  bf16* vtbf = (bf16*)(ws + off); off += XB;   // V^T per head [(b*H+h)*64+d][s]
  bf16* abf  = (bf16*)(ws + off); off += XB;   // attention output, bf16

  const dim3 blk(256);

  // 1) fp32 -> bf16 conversions
  {
    const int nx4 = (MROWS * D_MODEL) / 4;
    const int nw4 = (D_MODEL * D_MODEL) / 4;
    cvt_f32_bf16<<<dim3(2048), blk, 0, stream>>>(x,  xbf,  nx4);
    cvt_f32_bf16<<<dim3(1024), blk, 0, stream>>>(Wq, wqbf, nw4);
    cvt_f32_bf16<<<dim3(1024), blk, 0, stream>>>(Wk, wkbf, nw4);
    cvt_f32_bf16<<<dim3(1024), blk, 0, stream>>>(Wv, wvbf, nw4);
    cvt_f32_bf16<<<dim3(1024), blk, 0, stream>>>(Wo, wobf, nw4);
  }

  // 2) Q/K/V projections (V written transposed per head)
  const dim3 ggrid(MROWS / 128, D_MODEL / 128);   // (64, 8)
  gemm_bf16<<<ggrid, blk, 0, stream>>>(xbf, wqbf, bq, (void*)qbf,  0);
  gemm_bf16<<<ggrid, blk, 0, stream>>>(xbf, wkbf, bk, (void*)kbf,  0);
  gemm_bf16<<<ggrid, blk, 0, stream>>>(xbf, wvbf, bv, (void*)vtbf, 1);

  // 3) causal flash attention (TDM-staged K/V tiles)
  attn_fwd<<<dim3(S_LEN / 128, BATCH * NHEADS), blk, 0, stream>>>(qbf, kbf, vtbf, abf);

  // 4) output projection, fp32 result
  gemm_bf16<<<ggrid, blk, 0, stream>>>(abf, wobf, bo, d_out, 2);
}
